// CrossAttention_14998025798116
// MI455X (gfx1250) — compile-verified, matrix operand order checked by reading, not run on hardware
//
#include <hip/hip_runtime.h>
#include <stdint.h>

#define BB 16
#define TT 2048
#define HH 16
#define DD 64
#define MD 1024
#define HID 4096
#define NTOK (BB * TT)

typedef __attribute__((ext_vector_type(8)))  float        v8f;
typedef __attribute__((ext_vector_type(16))) __bf16       v16bf;
typedef __attribute__((ext_vector_type(4)))  unsigned int v4u;
typedef __attribute__((ext_vector_type(8)))  unsigned int v8u;
typedef __attribute__((ext_vector_type(4)))  int          v4i;
typedef __attribute__((address_space(1))) v4i gas_v4i;  // global int4
typedef __attribute__((address_space(3))) v4i las_v4i;  // LDS int4

#ifndef __has_builtin
#define __has_builtin(x) 0
#endif
#if __has_builtin(__builtin_amdgcn_global_load_async_to_lds_b128) && \
    __has_builtin(__builtin_amdgcn_s_wait_asynccnt)
#define HAVE_ASYNC_LDS 1
#else
#define HAVE_ASYNC_LDS 0
#endif

#if HAVE_ASYNC_LDS
#define ASYNC_CP16(gsrc, ldst)                                  \
  __builtin_amdgcn_global_load_async_to_lds_b128(               \
      (gas_v4i*)(gsrc), (las_v4i*)(ldst), 0, 0)
#endif

__device__ __forceinline__ unsigned short f2bf(float f) {
  union { float f; unsigned u; } x; x.f = f;
  unsigned r = x.u + 0x7FFFu + ((x.u >> 16) & 1u);   // round-to-nearest-even
  return (unsigned short)(r >> 16);
}

// Build a 16-element bf16 WMMA fragment from two 16-byte LDS chunks.
// CDNA5 16-bit A/B layout: lane<16 holds K{0..7 | 16..23},
// lane>=16 holds K{8..15 | 24..31} (caller passes chunk bases).
__device__ __forceinline__ v16bf frag16(const unsigned short* p0,
                                        const unsigned short* p1) {
  v4u lo = *(const v4u*)p0;
  v4u hi = *(const v4u*)p1;
  v8u u  = __builtin_shufflevector(lo, hi, 0, 1, 2, 3, 4, 5, 6, 7);
  return __builtin_bit_cast(v16bf, u);
}

#define WMMA_BF16(a, b, c)                                                  \
  __builtin_amdgcn_wmma_f32_16x16x32_bf16(false, (a), false, (b), (short)0, \
                                          (c), false, false)

// ---------------------------------------------------------------------------
// Tiled transpose + f32->bf16:  dst[c*R + r] = bf16(src[r*C + c])
// grid = (C/32, R/32), block = 256
// ---------------------------------------------------------------------------
__global__ __launch_bounds__(256) void xattn_cvt_transpose(
    const float* __restrict__ src, unsigned short* __restrict__ dst,
    int R, int C) {
  __shared__ float tile[32][33];
  const int tx = threadIdx.x & 31;
  const int ty = threadIdx.x >> 5;  // 0..7
  const int c0 = blockIdx.x * 32;
  const int r0 = blockIdx.y * 32;
#pragma unroll
  for (int p = 0; p < 4; ++p)
    tile[ty + p * 8][tx] = src[(size_t)(r0 + ty + p * 8) * C + c0 + tx];
  __syncthreads();
#pragma unroll
  for (int p = 0; p < 4; ++p)
    dst[(size_t)(c0 + ty + p * 8) * R + r0 + tx] = f2bf(tile[tx][ty + p * 8]);
}

// ---------------------------------------------------------------------------
// RMSNorm one token per block -> bf16 (vectorized)
// ---------------------------------------------------------------------------
__global__ __launch_bounds__(256) void xattn_rmsnorm(
    const float* __restrict__ x, const float* __restrict__ w,
    unsigned short* __restrict__ xn) {
  __shared__ float red[256];
  const int tid = threadIdx.x;
  const size_t base = (size_t)blockIdx.x * MD;
  float4 x4 = *(const float4*)(x + base + tid * 4);
  red[tid] = x4.x * x4.x + x4.y * x4.y + x4.z * x4.z + x4.w * x4.w;
  __syncthreads();
  for (int s = 128; s > 0; s >>= 1) {
    if (tid < s) red[tid] += red[tid + s];
    __syncthreads();
  }
  const float k = rsqrtf(red[0] * (1.0f / MD) + 1e-8f);
  float4 w4 = *(const float4*)(w + tid * 4);
  uint2 o;
  o.x = (unsigned)f2bf(x4.x * k * w4.x) | ((unsigned)f2bf(x4.y * k * w4.y) << 16);
  o.y = (unsigned)f2bf(x4.z * k * w4.z) | ((unsigned)f2bf(x4.w * k * w4.w) << 16);
  *(uint2*)(xn + base + tid * 4) = o;
}

// ---------------------------------------------------------------------------
// FFN up: hid = silu(xn@Wg + bg) * (xn@Wv + bv), bf16 out.
// Tile 64(M) x 64(N), K-step 32, double-buffered LDS with async global->LDS
// staging (ASYNCcnt), 8 waves (4Mx2N): 1 A-frag -> 4 WMMA per step.
// ---------------------------------------------------------------------------
__global__ __launch_bounds__(256) void xattn_ffn_up(
    const unsigned short* __restrict__ xn,
    const unsigned short* __restrict__ wgT,  // [HID, MD] bf16
    const unsigned short* __restrict__ wvT,  // [HID, MD] bf16
    const float* __restrict__ bgp, const float* __restrict__ bvp,
    unsigned short* __restrict__ hid) {
  __shared__ __align__(16) unsigned short As[2][64][40];
  __shared__ __align__(16) unsigned short Bg[2][64][40];
  __shared__ __align__(16) unsigned short Bv[2][64][40];
  const int tid  = threadIdx.x;
  const int lane = tid & 31, wave = tid >> 5;
  const int wm = wave & 3, wn = wave >> 2;
  const int row0 = blockIdx.y * 64;
  const int col0 = blockIdx.x * 64;

  // staging: each thread owns one 16B chunk per 64x32 tile
  const int sr = tid >> 2;        // 0..63
  const int sc = (tid & 3) * 8;   // k element offset
  const unsigned short* ga = xn  + (size_t)(row0 + sr) * MD + sc;
  const unsigned short* gg = wgT + (size_t)(col0 + sr) * MD + sc;
  const unsigned short* gv = wvT + (size_t)(col0 + sr) * MD + sc;

  v8f cg[2] = {};
  v8f cv[2] = {};

#if HAVE_ASYNC_LDS
  ASYNC_CP16(ga, &As[0][sr][sc]);
  ASYNC_CP16(gg, &Bg[0][sr][sc]);
  ASYNC_CP16(gv, &Bv[0][sr][sc]);
  __builtin_amdgcn_s_wait_asynccnt(0);
#else
  *(v4u*)&As[0][sr][sc] = *(const v4u*)ga;
  *(v4u*)&Bg[0][sr][sc] = *(const v4u*)gg;
  *(v4u*)&Bv[0][sr][sc] = *(const v4u*)gv;
#endif
  __syncthreads();

  const int mr = wm * 16 + (lane & 15);
  const int kh = (lane >> 4) * 8;
  const int NSTEP = MD / 32;

  for (int s = 0; s < NSTEP; ++s) {
    const int cur = s & 1;
    const bool more = (s + 1) < NSTEP;
#if HAVE_ASYNC_LDS
    if (more) {
      const int ko = (s + 1) * 32;
      ASYNC_CP16(ga + ko, &As[1 - cur][sr][sc]);
      ASYNC_CP16(gg + ko, &Bg[1 - cur][sr][sc]);
      ASYNC_CP16(gv + ko, &Bv[1 - cur][sr][sc]);
    }
#else
    v4u ra, rg, rv;
    if (more) {
      const int ko = (s + 1) * 32;
      ra = *(const v4u*)(ga + ko);
      rg = *(const v4u*)(gg + ko);
      rv = *(const v4u*)(gv + ko);
    }
#endif
    if (s + 2 < NSTEP) __builtin_prefetch(ga + (s + 2) * 32, 0, 2);
    v16bf a = frag16(&As[cur][mr][kh], &As[cur][mr][16 + kh]);
#pragma unroll
    for (int j = 0; j < 2; ++j) {
      const int nr = wn * 32 + j * 16 + (lane & 15);
      v16bf bg = frag16(&Bg[cur][nr][kh], &Bg[cur][nr][16 + kh]);
      v16bf bv = frag16(&Bv[cur][nr][kh], &Bv[cur][nr][16 + kh]);
      cg[j] = WMMA_BF16(a, bg, cg[j]);
      cv[j] = WMMA_BF16(a, bv, cv[j]);
    }
#if HAVE_ASYNC_LDS
    if (more) __builtin_amdgcn_s_wait_asynccnt(0);
#else
    if (more) {
      *(v4u*)&As[1 - cur][sr][sc] = ra;
      *(v4u*)&Bg[1 - cur][sr][sc] = rg;
      *(v4u*)&Bv[1 - cur][sr][sc] = rv;
    }
#endif
    __syncthreads();
  }

  // C layout: element i -> (m = i + 8*(lane>>4), n = lane&15)
  const int rowb = row0 + wm * 16 + 8 * (lane >> 4);
#pragma unroll
  for (int j = 0; j < 2; ++j) {
    const int n = col0 + wn * 32 + j * 16 + (lane & 15);
    const float bgn = bgp[n], bvn = bvp[n];
#pragma unroll
    for (int i = 0; i < 8; ++i) {
      float g  = cg[j][i] + bgn;
      float vv = cv[j][i] + bvn;
      float h  = (g / (1.f + __expf(-g))) * vv;  // silu(g)*v
      hid[(size_t)(rowb + i) * HID + n] = f2bf(h);
    }
  }
}

// ---------------------------------------------------------------------------
// FFN down: seq_hidden = hid@Wo + bo + residual (f32 out)
// Tile 64x64, double-buffered, async global->LDS staging (ASYNCcnt).
// ---------------------------------------------------------------------------
__global__ __launch_bounds__(256) void xattn_ffn_down(
    const unsigned short* __restrict__ hid,  // [NTOK, HID] bf16
    const unsigned short* __restrict__ woT,  // [MD, HID] bf16
    const float* __restrict__ bo,
    const float* __restrict__ resid,
    float* __restrict__ out) {
  __shared__ __align__(16) unsigned short As[2][64][40];
  __shared__ __align__(16) unsigned short Bs[2][64][40];
  const int tid  = threadIdx.x;
  const int lane = tid & 31, wave = tid >> 5;
  const int wm = wave & 3, wn = wave >> 2;
  const int row0 = blockIdx.y * 64;
  const int col0 = blockIdx.x * 64;

  const int sr = tid >> 2;
  const int sc = (tid & 3) * 8;
  const unsigned short* ga = hid + (size_t)(row0 + sr) * HID + sc;
  const unsigned short* gb = woT + (size_t)(col0 + sr) * HID + sc;

  v8f c[2] = {};

#if HAVE_ASYNC_LDS
  ASYNC_CP16(ga, &As[0][sr][sc]);
  ASYNC_CP16(gb, &Bs[0][sr][sc]);
  __builtin_amdgcn_s_wait_asynccnt(0);
#else
  *(v4u*)&As[0][sr][sc] = *(const v4u*)ga;
  *(v4u*)&Bs[0][sr][sc] = *(const v4u*)gb;
#endif
  __syncthreads();

  const int mr = wm * 16 + (lane & 15);
  const int kh = (lane >> 4) * 8;
  const int NSTEP = HID / 32;

  for (int s = 0; s < NSTEP; ++s) {
    const int cur = s & 1;
    const bool more = (s + 1) < NSTEP;
#if HAVE_ASYNC_LDS
    if (more) {
      const int ko = (s + 1) * 32;
      ASYNC_CP16(ga + ko, &As[1 - cur][sr][sc]);
      ASYNC_CP16(gb + ko, &Bs[1 - cur][sr][sc]);
    }
#else
    v4u ra, rb;
    if (more) {
      const int ko = (s + 1) * 32;
      ra = *(const v4u*)(ga + ko);
      rb = *(const v4u*)(gb + ko);
    }
#endif
    if (s + 2 < NSTEP) __builtin_prefetch(ga + (s + 2) * 32, 0, 2);
    v16bf a = frag16(&As[cur][mr][kh], &As[cur][mr][16 + kh]);
#pragma unroll
    for (int j = 0; j < 2; ++j) {
      const int nr = wn * 32 + j * 16 + (lane & 15);
      v16bf b = frag16(&Bs[cur][nr][kh], &Bs[cur][nr][16 + kh]);
      c[j] = WMMA_BF16(a, b, c[j]);
    }
#if HAVE_ASYNC_LDS
    if (more) __builtin_amdgcn_s_wait_asynccnt(0);
#else
    if (more) {
      *(v4u*)&As[1 - cur][sr][sc] = ra;
      *(v4u*)&Bs[1 - cur][sr][sc] = rb;
    }
#endif
    __syncthreads();
  }

  const int rowb = row0 + wm * 16 + 8 * (lane >> 4);
#pragma unroll
  for (int j = 0; j < 2; ++j) {
    const int n = col0 + wn * 32 + j * 16 + (lane & 15);
    const float bn = bo[n];
#pragma unroll
    for (int i = 0; i < 8; ++i) {
      size_t idx = (size_t)(rowb + i) * MD + n;
      out[idx] = c[j][i] + bn + resid[idx];
    }
  }
}

// ---------------------------------------------------------------------------
// Attention with folded projections:
//   qk = (K_h q)/sqrt(D);  score_t = x_t . qk + (q.k_b)/sqrt(D)
//   z   = (sum_t softmax_t * x_t) @ V_h + v_b + q
// One block per (b,h), 256 threads.
// ---------------------------------------------------------------------------
__global__ __launch_bounds__(256) void xattn_attn(
    const float* __restrict__ q, const float* __restrict__ kw,
    const float* __restrict__ kb, const float* __restrict__ vw,
    const float* __restrict__ vb, const unsigned char* __restrict__ mask,
    const float* __restrict__ xh, float* __restrict__ z) {
  __shared__ float qk[DD];
  __shared__ float sc[TT];
  __shared__ float red[256];
  __shared__ float part[4][DD];
  __shared__ float qb_s;
  const int bh = blockIdx.x;
  const int b = bh / HH, h = bh % HH;
  const int tid = threadIdx.x;
  const float* qv = q + (size_t)bh * DD;

  if (tid < DD) {
    float acc = 0.f;
#pragma unroll 8
    for (int e = 0; e < DD; ++e)
      acc += kw[((size_t)h * DD + tid) * DD + e] * qv[e];
    qk[tid] = acc * 0.125f;  // 1/sqrt(64)
  }
  if (tid == 0) {
    float a = 0.f;
    for (int e = 0; e < DD; ++e) a += kb[h * DD + e] * qv[e];
    qb_s = a * 0.125f;
  }
  __syncthreads();
  const float qb = qb_s;

  float lmax = -3.4e38f;
#pragma unroll
  for (int i = 0; i < TT / 256; ++i) {
    const int t = tid * (TT / 256) + i;
    const float4* xr = (const float4*)(xh + ((size_t)(b * TT + t) * MD + h * DD));
    float acc = 0.f;
#pragma unroll
    for (int d4 = 0; d4 < DD / 4; ++d4) {
      float4 x4 = xr[d4];
      acc += x4.x * qk[d4 * 4] + x4.y * qk[d4 * 4 + 1] +
             x4.z * qk[d4 * 4 + 2] + x4.w * qk[d4 * 4 + 3];
    }
    float s = acc + qb;
    if (!mask[b * TT + t]) s = -3.4e38f;
    sc[t] = s;
    lmax = fmaxf(lmax, s);
  }
  red[tid] = lmax;
  __syncthreads();
  for (int s = 128; s > 0; s >>= 1) {
    if (tid < s) red[tid] = fmaxf(red[tid], red[tid + s]);
    __syncthreads();
  }
  const float gmax = red[0];
  __syncthreads();

  float lsum = 0.f;
#pragma unroll
  for (int i = 0; i < TT / 256; ++i) {
    const int t = tid * (TT / 256) + i;
    float e = __expf(sc[t] - gmax);
    sc[t] = e;
    lsum += e;
  }
  red[tid] = lsum;
  __syncthreads();
  for (int s = 128; s > 0; s >>= 1) {
    if (tid < s) red[tid] += red[tid + s];
    __syncthreads();
  }
  const float inv = 1.f / red[0];
  __syncthreads();

  const int dg = tid & 63, grp = tid >> 6;
  float acc = 0.f;
  for (int t = grp * (TT / 4); t < (grp + 1) * (TT / 4); ++t)
    acc += sc[t] * xh[(size_t)(b * TT + t) * MD + h * DD + dg];
  part[grp][dg] = acc;
  __syncthreads();
  if (tid < DD) {
    float sd = (part[0][tid] + part[1][tid] + part[2][tid] + part[3][tid]) * inv;
    part[0][tid] = sd;
  }
  __syncthreads();
  if (tid < DD) {
    const int e = tid;
    float acc2 = 0.f;
#pragma unroll 8
    for (int d = 0; d < DD; ++d)
      acc2 += part[0][d] * vw[((size_t)h * DD + d) * DD + e];
    z[(size_t)bh * DD + e] = acc2 + vb[h * DD + e] + qv[e];
  }
}

// ---------------------------------------------------------------------------
extern "C" void kernel_launch(void* const* d_in, const int* in_sizes, int n_in,
                              void* d_out, int out_size, void* d_ws,
                              size_t ws_size, hipStream_t stream) {
  const float* q          = (const float*)d_in[0];
  const float* seq_repr   = (const float*)d_in[1];
  const unsigned char* mk = (const unsigned char*)d_in[2];
  const float* nw         = (const float*)d_in[3];
  const float* w_gate     = (const float*)d_in[4];
  const float* b_gate     = (const float*)d_in[5];
  const float* w_val      = (const float*)d_in[6];
  const float* b_val      = (const float*)d_in[7];
  const float* w_out      = (const float*)d_in[8];
  const float* b_out      = (const float*)d_in[9];
  const float* k_w        = (const float*)d_in[10];
  const float* k_b        = (const float*)d_in[11];
  const float* v_w        = (const float*)d_in[12];
  const float* v_b        = (const float*)d_in[13];

  float* z_out      = (float*)d_out;                         // [B,H,D]
  float* seq_hidden = (float*)d_out + (size_t)BB * HH * DD;  // [B,T,MD]

  char* ws = (char*)d_ws;
  unsigned short* xn  = (unsigned short*)ws;                            // NTOK*MD
  unsigned short* hid = (unsigned short*)(ws + (size_t)NTOK * MD * 2);  // NTOK*HID
  unsigned short* wgT = (unsigned short*)(ws + (size_t)NTOK * MD * 2 +
                                          (size_t)NTOK * HID * 2);
  unsigned short* wvT = wgT + (size_t)MD * HID;
  unsigned short* woT = wvT + (size_t)MD * HID;

  xattn_cvt_transpose<<<dim3(HID / 32, MD / 32), 256, 0, stream>>>(w_gate, wgT, MD, HID);
  xattn_cvt_transpose<<<dim3(HID / 32, MD / 32), 256, 0, stream>>>(w_val, wvT, MD, HID);
  xattn_cvt_transpose<<<dim3(MD / 32, HID / 32), 256, 0, stream>>>(w_out, woT, HID, MD);
  xattn_rmsnorm<<<NTOK, 256, 0, stream>>>(seq_repr, nw, xn);
  xattn_ffn_up<<<dim3(HID / 64, NTOK / 64), 256, 0, stream>>>(
      xn, wgT, wvT, b_gate, b_val, hid);
  xattn_ffn_down<<<dim3(MD / 64, NTOK / 64), 256, 0, stream>>>(
      hid, woT, b_out, seq_repr, seq_hidden);
  xattn_attn<<<BB * HH, 256, 0, stream>>>(q, k_w, k_b, v_w, v_b, mk,
                                          seq_hidden, z_out);
}
